// ObjectMiningOutputLayers_68083821576579
// MI455X (gfx1250) — matrix operand order repaired and checked by reading, not compiled, not run on hardware
//
#include <hip/hip_runtime.h>
#include <hip/hip_bf16.h>

typedef __attribute__((ext_vector_type(16))) __bf16 v16bf;
typedef __attribute__((ext_vector_type(2)))  __bf16 v2bf;
typedef __attribute__((ext_vector_type(8)))  float  v8f;
typedef __attribute__((ext_vector_type(4)))  int    v4i;

#define R_DIM 8192
#define D_DIM 4096
#define C_DIM 80
#define K_PRE 512
#define TOPK  100

#if defined(__HIP_DEVICE_COMPILE__) && __has_builtin(__builtin_amdgcn_global_load_async_to_lds_b128)
#define USE_ASYNC_LDS 1
typedef __attribute__((address_space(1))) v4i gv4i;   // global int4*
typedef __attribute__((address_space(3))) v4i sv4i;   // LDS int4*
#else
#define USE_ASYNC_LDS 0
#endif

__device__ __forceinline__ unsigned short f32_to_bf16(float f) {
    unsigned int u = __float_as_uint(f);
    unsigned int lsb = (u >> 16) & 1u;
    u += 0x7fffu + lsb;                      // round-to-nearest-even
    return (unsigned short)(u >> 16);
}

// ---------------------------------------------------------------------------
// 1. Pack Wc/Wd (fp32, [4096,80]) into bf16, B-operand lane layout in contiguous k order,
//    arranged [kt][head][nt][512] so each K-step's 10 tiles are one
//    contiguous 10240B block (async-staged into LDS per workgroup).
//    Within a 32x16 tile, element (ko,no):
//      lane = (ko/16)*16 + no ; vgpr = (ko%16)/2 ; pos = ko%2
// ---------------------------------------------------------------------------
__global__ void pack_weights(const float* __restrict__ Wc,
                             const float* __restrict__ Wd,
                             unsigned short* __restrict__ wp) {
    int i = blockIdx.x * blockDim.x + threadIdx.x;
    if (i >= D_DIM * C_DIM) return;
    int k = i / C_DIM, n = i % C_DIM;
    int kt = k >> 5, ko = k & 31;
    int nt = n >> 4, no = n & 15;
    int lane = ((ko >> 4) << 4) + no;
    int v = (ko & 15) >> 1, pos = ko & 1;
    size_t in_tile = (size_t)lane * 16 + v * 2 + pos;
    wp[((size_t)kt * 10 + 0 * 5 + nt) * 512 + in_tile] = f32_to_bf16(Wc[i]);
    wp[((size_t)kt * 10 + 1 * 5 + nt) * 512 + in_tile] = f32_to_bf16(Wd[i]);
}

// ---------------------------------------------------------------------------
// 2. Fused dual-head GEMM: cls = x@Wc + bc ; det = x@Wd + bd  via bf16 WMMA.
//    64 WGs x 256 threads (8 waves). Wave handles 16 rows x 80 cols.
//    Both the x tile (128x32 fp32) and the B tiles (10KB) are double-
//    buffered in LDS and fetched with async-to-LDS while WMMAs run.
// ---------------------------------------------------------------------------
#define XS_STRIDE 36   // floats; 36*4=144B -> 16B-aligned rows, conflict-free

__global__ __launch_bounds__(256) void gemm_wmma(
        const float* __restrict__ x,
        const unsigned short* __restrict__ wp,
        const float* __restrict__ bc, const float* __restrict__ bd,
        float* __restrict__ cls, float* __restrict__ det) {
    __shared__ __align__(16) float          xs[2][128][XS_STRIDE]; // 36.9 KB
    __shared__ __align__(16) unsigned short bs[2][5120];           // 20.5 KB
    const int tid   = threadIdx.x;
    const int wave  = tid >> 5;
    const int lane  = tid & 31;
    const int half  = lane >> 4;
    const int mlane = lane & 15;
    const int rowBase = blockIdx.x * 128;

    v8f acc_c[5] = {};
    v8f acc_d[5] = {};

    // stage one 128x32 fp32 tile of x into LDS buffer p
    auto stage_x = [&](int kt, int p) {
        const int kb = kt << 5;
        #pragma unroll
        for (int it = 0; it < 4; ++it) {
            int idx = tid + it * 256;        // 1024 float4 total
            int r  = idx >> 3;
            int c4 = idx & 7;
            const float* gp = x + (size_t)(rowBase + r) * D_DIM + kb + c4 * 4;
            float* lp = &xs[p][r][c4 * 4];
#if USE_ASYNC_LDS
            __builtin_amdgcn_global_load_async_to_lds_b128(
                (gv4i*)gp, (sv4i*)lp, 0, 0);
#else
            const float4 f = *reinterpret_cast<const float4*>(gp);
            lp[0] = f.x; lp[1] = f.y; lp[2] = f.z; lp[3] = f.w;
            __builtin_prefetch(gp + 32, 0, 1);
#endif
        }
    };

    // stage this K-step's 10 B tiles (10240 bytes) into LDS buffer p
    auto stage_b = [&](int kt, int p) {
        const unsigned short* src = wp + (size_t)kt * 5120;
        #pragma unroll
        for (int it = 0; it < 3; ++it) {
            int i = tid + it * 256;          // 640 x 16B chunks
            if (i < 640) {
#if USE_ASYNC_LDS
                __builtin_amdgcn_global_load_async_to_lds_b128(
                    (gv4i*)(src + i * 8), (sv4i*)(&bs[p][i * 8]), 0, 0);
#else
                v4i t = *reinterpret_cast<const v4i*>(src + i * 8);
                *reinterpret_cast<v4i*>(&bs[p][i * 8]) = t;
#endif
            }
        }
    };

    stage_x(0, 0);
    stage_b(0, 0);
    int p = 0;

    for (int kt = 0; kt < 128; ++kt) {
#if USE_ASYNC_LDS
#if __has_builtin(__builtin_amdgcn_s_wait_asynccnt)
        __builtin_amdgcn_s_wait_asynccnt(0);
#else
        asm volatile("s_wait_asynccnt 0" ::: "memory");
#endif
#endif
        __syncthreads();                     // tile kt ready in xs[p]/bs[p]
        if (kt + 1 < 128) { stage_x(kt + 1, p ^ 1); stage_b(kt + 1, p ^ 1); }

        // build A fragment: contiguous k order, k = 16*half + element (matches packed B)
        // lanes 0-15: v0..7 -> K=2v,2v+1 ; lanes 16-31: K=16+2v,16+2v+1
        const float* xrow = &xs[p][wave * 16 + mlane][0];
        v16bf a;
        #pragma unroll
        for (int v = 0; v < 8; ++v) {
            int kp = 2 * v + half * 16;
            float f0 = xrow[kp];
            float f1 = xrow[kp + 1];
#if __has_builtin(__builtin_amdgcn_cvt_pk_bf16_f32)
            v2bf pk = __builtin_amdgcn_cvt_pk_bf16_f32(f0, f1);
            a[2 * v]     = pk[0];
            a[2 * v + 1] = pk[1];
#else
            a[2 * v]     = __builtin_bit_cast(__bf16, f32_to_bf16(f0));
            a[2 * v + 1] = __builtin_bit_cast(__bf16, f32_to_bf16(f1));
#endif
        }

        // B fragments now come from LDS (cheap ds_load_b128 pairs)
        const unsigned short* bbase = &bs[p][0];
        v16bf bC[5], bD[5];
        #pragma unroll
        for (int nt = 0; nt < 5; ++nt) {
            bC[nt] = *reinterpret_cast<const v16bf*>(bbase + (size_t)nt * 512 + lane * 16);
            bD[nt] = *reinterpret_cast<const v16bf*>(bbase + (size_t)(5 + nt) * 512 + lane * 16);
        }
        #pragma unroll
        for (int nt = 0; nt < 5; ++nt) {
            acc_c[nt] = __builtin_amdgcn_wmma_f32_16x16x32_bf16(
                false, a, false, bC[nt], (short)0, acc_c[nt], false, false);
            acc_d[nt] = __builtin_amdgcn_wmma_f32_16x16x32_bf16(
                false, a, false, bD[nt], (short)0, acc_d[nt], false, false);
        }
        p ^= 1;
    }

    // epilogue: C/D layout — VGPR r: lanes0-15 M=r, lanes16-31 M=8+r; N=lane%16
    #pragma unroll
    for (int nt = 0; nt < 5; ++nt) {
        #pragma unroll
        for (int r = 0; r < 8; ++r) {
            int m   = (half << 3) + r;
            int row = rowBase + wave * 16 + m;
            int col = nt * 16 + mlane;
            cls[(size_t)row * C_DIM + col] = acc_c[nt][r] + bc[col];
            det[(size_t)row * C_DIM + col] = acc_d[nt][r] + bd[col];
        }
    }
}

// ---------------------------------------------------------------------------
// 3. Column softmax stats for det_logits (max + sumexp over 8192 rows/class)
// ---------------------------------------------------------------------------
__global__ void col_stats(const float* __restrict__ det,
                          float* __restrict__ col_max,
                          float* __restrict__ col_sum) {
    __shared__ float red[256];
    __shared__ float bmax;
    const int c = blockIdx.x, tid = threadIdx.x;
    float m = -3.4e38f;
    for (int r = tid; r < R_DIM; r += 256)
        m = fmaxf(m, det[(size_t)r * C_DIM + c]);
    red[tid] = m; __syncthreads();
    for (int s = 128; s > 0; s >>= 1) {
        if (tid < s) red[tid] = fmaxf(red[tid], red[tid + s]);
        __syncthreads();
    }
    if (tid == 0) { bmax = red[0]; col_max[c] = red[0]; }
    __syncthreads();
    const float mm = bmax;
    float sum = 0.f;
    for (int r = tid; r < R_DIM; r += 256)
        sum += __expf(det[(size_t)r * C_DIM + c] - mm);
    red[tid] = sum; __syncthreads();
    for (int s = 128; s > 0; s >>= 1) {
        if (tid < s) red[tid] += red[tid + s];
        __syncthreads();
    }
    if (tid == 0) col_sum[c] = red[0];
}

// ---------------------------------------------------------------------------
// 4. pred = softmax_row(cls) * softmax_col(det)
// ---------------------------------------------------------------------------
__global__ void pred_kernel(const float* __restrict__ cls,
                            const float* __restrict__ det,
                            const float* __restrict__ col_max,
                            const float* __restrict__ col_sum,
                            float* __restrict__ pred) {
    __shared__ float cm[C_DIM], cs[C_DIM];
    const int tid = threadIdx.x;
    if (tid < C_DIM) { cm[tid] = col_max[tid]; cs[tid] = col_sum[tid]; }
    __syncthreads();
    const int row = blockIdx.x * blockDim.x + tid;
    if (row >= R_DIM) return;
    const float* cr = cls + (size_t)row * C_DIM;
    const float* dr = det + (size_t)row * C_DIM;
    float rmax = -3.4e38f;
    for (int c = 0; c < C_DIM; ++c) rmax = fmaxf(rmax, cr[c]);
    float rsum = 0.f;
    for (int c = 0; c < C_DIM; ++c) rsum += __expf(cr[c] - rmax);
    const float inv = 1.f / rsum;
    for (int c = 0; c < C_DIM; ++c) {
        float pc = __expf(cr[c] - rmax) * inv;
        float pd = __expf(dr[c] - cm[c]) / cs[c];
        pred[(size_t)row * C_DIM + c] = pc * pd;
    }
}

// ---------------------------------------------------------------------------
// 5. Per-class top-512 (in-LDS bitonic sort, descending) + greedy NMS.
//    One workgroup per class; writes kept scores (others 0) into out.
// ---------------------------------------------------------------------------
__global__ __launch_bounds__(1024) void nms_kernel(
        const float* __restrict__ pred,
        const float* __restrict__ boxes,
        float* __restrict__ out) {
    __shared__ float          s[R_DIM];     // 32 KB
    __shared__ unsigned short id[R_DIM];    // 16 KB
    __shared__ float          kb[K_PRE][4]; //  8 KB
    __shared__ int            kept[K_PRE];  //  2 KB
    __shared__ int            flag;
    const int c = blockIdx.x, tid = threadIdx.x;

    for (int r = tid; r < R_DIM; r += 1024) {
        s[r]  = pred[(size_t)r * C_DIM + c];
        id[r] = (unsigned short)r;
    }
    __syncthreads();

    // bitonic sort, descending
    for (int k = 2; k <= R_DIM; k <<= 1) {
        for (int j = k >> 1; j > 0; j >>= 1) {
            for (int t = tid; t < R_DIM; t += 1024) {
                int ixj = t ^ j;
                if (ixj > t) {
                    bool up = ((t & k) == 0);        // descending segment
                    float si = s[t], sj = s[ixj];
                    bool sw = up ? (si < sj) : (si > sj);
                    if (sw) {
                        s[t] = sj; s[ixj] = si;
                        unsigned short tmp = id[t]; id[t] = id[ixj]; id[ixj] = tmp;
                    }
                }
            }
            __syncthreads();
        }
    }

    // clip boxes of top-512 into LDS
    if (tid < K_PRE) {
        int r = id[tid];
        kb[tid][0] = fminf(fmaxf(boxes[r * 4 + 0], 0.f), 1000.f);
        kb[tid][1] = fminf(fmaxf(boxes[r * 4 + 1], 0.f),  800.f);
        kb[tid][2] = fminf(fmaxf(boxes[r * 4 + 2], 0.f), 1000.f);
        kb[tid][3] = fminf(fmaxf(boxes[r * 4 + 3], 0.f),  800.f);
        kept[tid] = 0;
    }
    __syncthreads();

    // greedy suppression, score order
    for (int i = 0; i < K_PRE; ++i) {
        if (tid == 0) flag = 0;
        __syncthreads();
        if (tid < i && kept[tid]) {
            float ax1 = kb[i][0], ay1 = kb[i][1], ax2 = kb[i][2], ay2 = kb[i][3];
            float bx1 = kb[tid][0], by1 = kb[tid][1], bx2 = kb[tid][2], by2 = kb[tid][3];
            float areaA = (ax2 - ax1) * (ay2 - ay1);
            float areaB = (bx2 - bx1) * (by2 - by1);
            float lx = fmaxf(ax1, bx1), ly = fmaxf(ay1, by1);
            float rx = fminf(ax2, bx2), ry = fminf(ay2, by2);
            float w = fmaxf(rx - lx, 0.f), h = fmaxf(ry - ly, 0.f);
            float inter = w * h;
            float iou = inter / (areaA + areaB - inter + 1e-9f);
            if (iou > 0.5f) flag = 1;        // benign race: only writes 1
        }
        __syncthreads();
        if (tid == 0) kept[i] = (s[i] > 1e-5f) && (flag == 0);
        __syncthreads();
    }

    for (int r = tid; r < R_DIM; r += 1024)
        out[(size_t)r * C_DIM + c] = 0.f;
    __syncthreads();
    if (tid < K_PRE && kept[tid])
        out[(size_t)id[tid] * C_DIM + c] = s[tid];
}

// ---------------------------------------------------------------------------
// 6. Exact 100th-largest kept score via binary search on float bit patterns.
// ---------------------------------------------------------------------------
__global__ __launch_bounds__(1024) void find_kth(const float* __restrict__ scores,
                                                 float* __restrict__ kth) {
    __shared__ int red[1024];
    const int tid = threadIdx.x;
    unsigned int best = 0u;
    for (int b = 30; b >= 0; --b) {
        unsigned int trial = best | (1u << b);
        int cnt = 0;
        for (int i = tid; i < R_DIM * C_DIM; i += 1024) {
            unsigned int vb = __float_as_uint(scores[i]);   // scores >= 0
            cnt += (vb >= trial) ? 1 : 0;
        }
        red[tid] = cnt; __syncthreads();
        for (int ss = 512; ss > 0; ss >>= 1) {
            if (tid < ss) red[tid] += red[tid + ss];
            __syncthreads();
        }
        int total = red[0];
        __syncthreads();
        if (total >= TOPK) best = trial;
    }
    if (tid == 0) *kth = __uint_as_float(best);
}

__global__ void apply_thresh(float* __restrict__ out, const float* __restrict__ kth) {
    int i = blockIdx.x * blockDim.x + threadIdx.x;
    if (i < R_DIM * C_DIM) {
        float v = out[i];
        out[i] = (v >= *kth) ? v : 0.f;
    }
}

// ---------------------------------------------------------------------------
extern "C" void kernel_launch(void* const* d_in, const int* in_sizes, int n_in,
                              void* d_out, int out_size, void* d_ws, size_t ws_size,
                              hipStream_t stream) {
    (void)in_sizes; (void)n_in; (void)out_size; (void)ws_size;
    const float* x     = (const float*)d_in[0];
    const float* Wc    = (const float*)d_in[1];
    const float* bc    = (const float*)d_in[2];
    const float* Wd    = (const float*)d_in[3];
    const float* bd    = (const float*)d_in[4];
    const float* boxes = (const float*)d_in[5];
    float* out = (float*)d_out;

    char* ws = (char*)d_ws;
    size_t off = 0;
    unsigned short* wp = (unsigned short*)(ws + off); off += (size_t)D_DIM * C_DIM * 2 * 2;
    float* cls     = (float*)(ws + off); off += (size_t)R_DIM * C_DIM * 4;
    float* det     = (float*)(ws + off); off += (size_t)R_DIM * C_DIM * 4;
    float* pred    = (float*)(ws + off); off += (size_t)R_DIM * C_DIM * 4;
    float* col_max = (float*)(ws + off); off += C_DIM * 4;
    float* col_sum = (float*)(ws + off); off += C_DIM * 4;
    float* kth     = (float*)(ws + off); off += 4;

    pack_weights<<<(D_DIM * C_DIM + 255) / 256, 256, 0, stream>>>(Wc, Wd, wp);
    gemm_wmma<<<R_DIM / 128, 256, 0, stream>>>(x, wp, bc, bd, cls, det);
    col_stats<<<C_DIM, 256, 0, stream>>>(det, col_max, col_sum);
    pred_kernel<<<R_DIM / 256, 256, 0, stream>>>(cls, det, col_max, col_sum, pred);
    nms_kernel<<<C_DIM, 1024, 0, stream>>>(pred, boxes, out);
    find_kth<<<1, 1024, 0, stream>>>(out, kth);
    apply_thresh<<<(R_DIM * C_DIM + 255) / 256, 256, 0, stream>>>(out, kth);
}
